// MLPBlock_7198365188469
// MI455X (gfx1250) — matrix operand order repaired and checked by reading, not compiled, not run on hardware
//
#include <hip/hip_runtime.h>

// MoE SwiGLU block for MI455X (gfx1250).
// Pipeline: [convert W1,W2 -> bf16] -> [RMSNorm+gate] -> [GEMM1+SwiGLU] -> [GEMM2+combine]
// GEMMs: v_wmma_f32_16x16x32_bf16; operands double-buffered in LDS via
// global_load_async_to_lds_b128 (ASYNCcnt), next tile prefetched under the WMMAs.
// ws layout: [ normed bf16 T*H | gate f32 T*E | act bf16 E*T*I | w1 bf16 | w2 bf16 ] (~232 MB)

typedef __attribute__((ext_vector_type(16))) __bf16       v16bf;
typedef __attribute__((ext_vector_type(8)))  float        v8f;
typedef __attribute__((ext_vector_type(4)))  unsigned int u32x4;
typedef __attribute__((ext_vector_type(4)))  float        f32x4;

constexpr int T = 4096, H = 1024, E = 8, I = 2048;
constexpr float EPS = 1e-5f;

__device__ __forceinline__ unsigned short f2bf(float f) {
  union { float f; unsigned int u; } v; v.f = f;
  unsigned int u = v.u;
  u += 0x7FFFu + ((u >> 16) & 1u);   // round-to-nearest-even
  return (unsigned short)(u >> 16);
}

union Frag { v16bf v; u32x4 q[2]; };

// gfx1250 async memory->LDS copy, 16B per lane, tracked by ASYNCcnt.
__device__ __forceinline__ void async_load_b128(void* lds_ptr, const void* gptr) {
  asm volatile("global_load_async_to_lds_b128 %0, %1, off"
               :: "v"((unsigned)(unsigned long long)lds_ptr),
                  "v"((unsigned long long)gptr)
               : "memory");
}
__device__ __forceinline__ void wait_async() {
  asm volatile("s_wait_asynccnt 0x0" ::: "memory");
}

// ---------------------------------------------------------------------------
// Kernel 0: one-shot fp32 -> bf16 weight conversion (memory-bound)
// ---------------------------------------------------------------------------
__global__ __launch_bounds__(256) void k_convert_bf16(
    const float* __restrict__ src, unsigned short* __restrict__ dst)
{
  const size_t idx = ((size_t)blockIdx.x * 256 + threadIdx.x) * 8;
  f32x4 a = *(const f32x4*)(src + idx);
  f32x4 b = *(const f32x4*)(src + idx + 4);
  u32x4 p;
  p[0] = (unsigned)f2bf(a[0]) | ((unsigned)f2bf(a[1]) << 16);
  p[1] = (unsigned)f2bf(a[2]) | ((unsigned)f2bf(a[3]) << 16);
  p[2] = (unsigned)f2bf(b[0]) | ((unsigned)f2bf(b[1]) << 16);
  p[3] = (unsigned)f2bf(b[2]) | ((unsigned)f2bf(b[3]) << 16);
  *(u32x4*)(dst + idx) = p;
}

// ---------------------------------------------------------------------------
// Kernel 1: RMSNorm (fp32) -> bf16 normed; gate logits + top-2 softmax -> gw
// ---------------------------------------------------------------------------
__global__ __launch_bounds__(256) void k_rmsnorm_gate(
    const float* __restrict__ x, const float* __restrict__ scale,
    const float* __restrict__ gk, const float* __restrict__ gb,
    unsigned short* __restrict__ normed, float* __restrict__ gw)
{
  __shared__ float red[256];
  __shared__ float gsum[256][E];
  const int t   = blockIdx.x;
  const int tid = threadIdx.x;

  const float* xr = x + (size_t)t * H;
  f32x4 xv = *(const f32x4*)(xr + tid * 4);
  red[tid] = xv[0]*xv[0] + xv[1]*xv[1] + xv[2]*xv[2] + xv[3]*xv[3];
  __syncthreads();
  for (int s = 128; s > 0; s >>= 1) {
    if (tid < s) red[tid] += red[tid + s];
    __syncthreads();
  }
  const float inv = rsqrtf(red[0] / (float)H + EPS);

  f32x4 sv = *(const f32x4*)(scale + tid * 4);
  float nf[4];
  unsigned short nh[4];
  #pragma unroll
  for (int j = 0; j < 4; ++j) { nf[j] = xv[j] * inv * sv[j]; nh[j] = f2bf(nf[j]); }
  unsigned int* np = (unsigned int*)(normed + (size_t)t * H + tid * 4);
  np[0] = (unsigned)nh[0] | ((unsigned)nh[1] << 16);
  np[1] = (unsigned)nh[2] | ((unsigned)nh[3] << 16);

  float acc[E];
  #pragma unroll
  for (int e = 0; e < E; ++e) acc[e] = 0.0f;
  const float* gkr = gk + (size_t)(tid * 4) * E;
  #pragma unroll
  for (int j = 0; j < 4; ++j)
    #pragma unroll
    for (int e = 0; e < E; ++e)
      acc[e] += nf[j] * gkr[j * E + e];
  #pragma unroll
  for (int e = 0; e < E; ++e) gsum[tid][e] = acc[e];
  __syncthreads();
  for (int s = 128; s > 0; s >>= 1) {
    if (tid < s)
      #pragma unroll
      for (int e = 0; e < E; ++e) gsum[tid][e] += gsum[tid + s][e];
    __syncthreads();
  }

  if (tid == 0) {
    float lg[E];
    #pragma unroll
    for (int e = 0; e < E; ++e) lg[e] = gsum[0][e] + gb[e];
    int i1 = 0;
    #pragma unroll
    for (int e = 1; e < E; ++e) if (lg[e] > lg[i1]) i1 = e;
    int i2 = (i1 == 0) ? 1 : 0;
    #pragma unroll
    for (int e = 0; e < E; ++e) if (e != i1 && lg[e] > lg[i2]) i2 = e;
    const float e2 = __expf(lg[i2] - lg[i1]);
    const float invs = 1.0f / (1.0f + e2);
    float* gwr = gw + (size_t)t * E;
    #pragma unroll
    for (int e = 0; e < E; ++e) gwr[e] = 0.0f;
    gwr[i1] = invs;
    gwr[i2] = e2 * invs;
  }
}

// ---------------------------------------------------------------------------
// Kernel 2: per-expert GEMM1 (bf16 WMMA) + fused SwiGLU -> bf16 act[E,T,I]
// Block tile: 64 tokens x 64 i-cols (gate AND linear halves). K-step 64,
// double-buffered LDS; async prefetch of next tile overlaps the WMMAs.
// ---------------------------------------------------------------------------
__global__ __launch_bounds__(256) void k_mlp1_swiglu(
    const unsigned short* __restrict__ normed,
    const unsigned short* __restrict__ w1b, const float* __restrict__ b1,
    unsigned short* __restrict__ act)
{
  __shared__ unsigned short As[2][64][72];
  __shared__ unsigned short Bg[2][64][72];
  __shared__ unsigned short Bl[2][64][72];

  const int t0 = blockIdx.x * 64;
  const int i0 = blockIdx.y * 64;
  const int e  = blockIdx.z;
  const int tid  = threadIdx.x;
  const int lane = tid & 31, wave = tid >> 5;
  const int m0  = (wave & 3) * 16;
  const int ng0 = (wave >> 2) * 32;

  const unsigned short* w1g = w1b + ((size_t)e * 2 * I + i0) * H;
  const unsigned short* w1l = w1b + ((size_t)e * 2 * I + I + i0) * H;

  v8f ag[2] = {};
  v8f al[2] = {};

  const int ar = tid >> 2, ac = (tid & 3) * 16;  // A: 16 bf16/thread (2 x b128)
  const int br = tid >> 1, bc = (tid & 1) * 32;  // B: 32 bf16/thread (4 x b128)
  const unsigned short* asrc0 = normed + (size_t)(t0 + ar) * H + ac;
  const unsigned short* bsrc0 = (br < 64 ? w1g + (size_t)br * H
                                         : w1l + (size_t)(br - 64) * H) + bc;
  unsigned short* bdst[2] = {
    (br < 64 ? &Bg[0][br][bc] : &Bl[0][br - 64][bc]),
    (br < 64 ? &Bg[1][br][bc] : &Bl[1][br - 64][bc]) };

  auto stage = [&](int buf, int kkArg) {
    async_load_b128(&As[buf][ar][ac],     asrc0 + kkArg);
    async_load_b128(&As[buf][ar][ac + 8], asrc0 + kkArg + 8);
    #pragma unroll
    for (int p = 0; p < 4; ++p)
      async_load_b128(bdst[buf] + p * 8, bsrc0 + kkArg + p * 8);
  };

  int cur = 0;
  stage(cur, 0);
  wait_async();
  __syncthreads();

  for (int kk = 0; kk < H; kk += 64) {
    const int nxt = cur ^ 1;
    if (kk + 64 < H) stage(nxt, kk + 64);   // prefetch under the WMMAs

    #pragma unroll
    for (int kc = 0; kc < 2; ++kc) {
      const int kb = kc * 32;
      // batch all fragment loads, then back-to-back WMMAs
      Frag a;
      const int am   = m0 + (lane & 15);
      const int koff = kb + (lane >> 4) * 16;
      a.q[0] = *(const u32x4*)&As[cur][am][koff];
      a.q[1] = *(const u32x4*)&As[cur][am][koff + 8];
      Frag bgf[2], blf[2];
      #pragma unroll
      for (int nt = 0; nt < 2; ++nt) {
        const int bn = ng0 + nt * 16 + (lane & 15);
        const int kh = kb + (lane >> 4) * 16;
        bgf[nt].q[0] = *(const u32x4*)&Bg[cur][bn][kh];
        bgf[nt].q[1] = *(const u32x4*)&Bg[cur][bn][kh + 8];
        blf[nt].q[0] = *(const u32x4*)&Bl[cur][bn][kh];
        blf[nt].q[1] = *(const u32x4*)&Bl[cur][bn][kh + 8];
      }
      #pragma unroll
      for (int nt = 0; nt < 2; ++nt) {
        ag[nt] = __builtin_amdgcn_wmma_f32_16x16x32_bf16(
            false, a.v, false, bgf[nt].v, (short)0, ag[nt], false, false);
        al[nt] = __builtin_amdgcn_wmma_f32_16x16x32_bf16(
            false, a.v, false, blf[nt].v, (short)0, al[nt], false, false);
      }
    }

    wait_async();
    __syncthreads();
    cur = nxt;
  }

  // epilogue: bias + clip + swish-gated linear, store bf16
  const int rsel = (lane >> 4) * 8;
  #pragma unroll
  for (int nt = 0; nt < 2; ++nt) {
    const int icol = i0 + ng0 + nt * 16 + (lane & 15);
    const float bgv = b1[(size_t)e * 2 * I + icol];
    const float blv = b1[(size_t)e * 2 * I + I + icol];
    #pragma unroll
    for (int j = 0; j < 8; ++j) {
      float g = ag[nt][j] + bgv;
      float l = al[nt][j] + blv;
      g = fminf(g, 7.0f);
      l = fminf(fmaxf(l, -7.0f), 7.0f);
      const float sw = g / (1.0f + __expf(-1.702f * g));
      const float av = sw * (l + 1.0f);
      const int trow = t0 + m0 + j + rsel;
      act[((size_t)e * T + trow) * I + icol] = f2bf(av);
    }
  }
}

// ---------------------------------------------------------------------------
// Kernel 3: per-expert GEMM2 (bf16 WMMA), gate-weighted combine + residual.
// Block tile: 64 tokens x 128 h. K-step 64, double-buffered LDS with async
// prefetch; loop all 8 experts with totals held in registers.
// ---------------------------------------------------------------------------
__global__ __launch_bounds__(256) void k_mlp2_combine(
    const unsigned short* __restrict__ act,
    const unsigned short* __restrict__ w2b, const float* __restrict__ b2,
    const float* __restrict__ gw, const float* __restrict__ x,
    float* __restrict__ out)
{
  __shared__ unsigned short As[2][64][72];
  __shared__ unsigned short Bs[2][128][72];
  __shared__ float gws[64][E];

  const int t0 = blockIdx.x * 64;
  const int h0 = blockIdx.y * 128;
  const int tid  = threadIdx.x;
  const int lane = tid & 31, wave = tid >> 5;
  const int m0 = (wave & 3) * 16;
  const int n0 = (wave >> 2) * 64;

  {
    const int idx = tid * 2;
    gws[idx / E][idx % E]             = gw[(size_t)(t0 + idx / E) * E + idx % E];
    gws[(idx + 1) / E][(idx + 1) % E] = gw[(size_t)(t0 + (idx + 1) / E) * E + (idx + 1) % E];
  }

  v8f tacc[4] = {};

  const int ar = tid >> 2, ac = (tid & 3) * 16;
  const int br = tid >> 1, bc = (tid & 1) * 32;

  for (int e = 0; e < E; ++e) {
    v8f eacc[4] = {};
    const unsigned short* aBase = act + ((size_t)e * T + t0 + ar) * I + ac;
    const unsigned short* bBase = w2b + ((size_t)e * H + h0 + br) * I + bc;

    auto stage = [&](int buf, int kkArg) {
      async_load_b128(&As[buf][ar][ac],     aBase + kkArg);
      async_load_b128(&As[buf][ar][ac + 8], aBase + kkArg + 8);
      #pragma unroll
      for (int p = 0; p < 4; ++p)
        async_load_b128(&Bs[buf][br][bc + p * 8], bBase + kkArg + p * 8);
    };

    int cur = 0;
    stage(cur, 0);
    wait_async();
    __syncthreads();

    for (int kk = 0; kk < I; kk += 64) {
      const int nxt = cur ^ 1;
      if (kk + 64 < I) stage(nxt, kk + 64);   // prefetch under the WMMAs

      #pragma unroll
      for (int kc = 0; kc < 2; ++kc) {
        const int kb = kc * 32;
        Frag a;
        const int am   = m0 + (lane & 15);
        const int koff = kb + (lane >> 4) * 16;
        a.q[0] = *(const u32x4*)&As[cur][am][koff];
        a.q[1] = *(const u32x4*)&As[cur][am][koff + 8];
        Frag bfr[4];
        #pragma unroll
        for (int nt = 0; nt < 4; ++nt) {
          const int bn = n0 + nt * 16 + (lane & 15);
          const int kh = kb + (lane >> 4) * 16;
          bfr[nt].q[0] = *(const u32x4*)&Bs[cur][bn][kh];
          bfr[nt].q[1] = *(const u32x4*)&Bs[cur][bn][kh + 8];
        }
        #pragma unroll
        for (int nt = 0; nt < 4; ++nt)
          eacc[nt] = __builtin_amdgcn_wmma_f32_16x16x32_bf16(
              false, a.v, false, bfr[nt].v, (short)0, eacc[nt], false, false);
      }

      wait_async();
      __syncthreads();
      cur = nxt;
    }

    const int rsel = (lane >> 4) * 8;
    #pragma unroll
    for (int nt = 0; nt < 4; ++nt) {
      const int hcol = h0 + n0 + nt * 16 + (lane & 15);
      const float b2v = b2[(size_t)e * H + hcol];
      #pragma unroll
      for (int j = 0; j < 8; ++j) {
        const float gwv = gws[m0 + j + rsel][e];
        tacc[nt][j] += gwv * (eacc[nt][j] + b2v);
      }
    }
  }

  const int rsel2 = (lane >> 4) * 8;
  #pragma unroll
  for (int nt = 0; nt < 4; ++nt) {
    const int hcol = h0 + n0 + nt * 16 + (lane & 15);
    #pragma unroll
    for (int j = 0; j < 8; ++j) {
      const int trow = t0 + m0 + j + rsel2;
      out[(size_t)trow * H + hcol] = x[(size_t)trow * H + hcol] + tacc[nt][j];
    }
  }
}

// ---------------------------------------------------------------------------
extern "C" void kernel_launch(void* const* d_in, const int* in_sizes, int n_in,
                              void* d_out, int out_size, void* d_ws, size_t ws_size,
                              hipStream_t stream) {
  const float* x     = (const float*)d_in[0];
  const float* scale = (const float*)d_in[1];
  const float* gk    = (const float*)d_in[2];
  const float* gb    = (const float*)d_in[3];
  const float* w1    = (const float*)d_in[4];
  const float* b1    = (const float*)d_in[5];
  const float* w2    = (const float*)d_in[6];
  const float* b2    = (const float*)d_in[7];
  float* out = (float*)d_out;

  // workspace carve-up
  char* ws = (char*)d_ws;
  size_t off = 0;
  unsigned short* normed = (unsigned short*)(ws + off); off += (size_t)T * H * 2;        // 8 MB
  float*          gwp    = (float*)         (ws + off); off += (size_t)T * E * 4;        // 128 KB
  unsigned short* actb   = (unsigned short*)(ws + off); off += (size_t)E * T * I * 2;    // 128 MB
  unsigned short* w1b    = (unsigned short*)(ws + off); off += (size_t)E * 2 * I * H * 2;// 64 MB
  unsigned short* w2b    = (unsigned short*)(ws + off);                                  // 32 MB

  const int n1 = E * 2 * I * H;   // 33,554,432
  const int n2 = E * H * I;       // 16,777,216
  k_convert_bf16<<<dim3(n1 / 2048), dim3(256), 0, stream>>>(w1, w1b);
  k_convert_bf16<<<dim3(n2 / 2048), dim3(256), 0, stream>>>(w2, w2b);
  k_rmsnorm_gate<<<dim3(T), dim3(256), 0, stream>>>(x, scale, gk, gb, normed, gwp);
  k_mlp1_swiglu<<<dim3(T / 64, I / 64, E), dim3(256), 0, stream>>>(normed, w1b, b1, actb);
  k_mlp2_combine<<<dim3(T / 64, H / 128), dim3(256), 0, stream>>>(actb, w2b, b2, gwp, x, out);
}
